// TactileGAT_82008105550327
// MI455X (gfx1250) — compile-verified, matrix operand order checked by reading, not run on hardware
//
#include <hip/hip_runtime.h>
#include <hip/hip_bf16.h>

// TactileGAT on gfx1250 (MI455X). Both matrix ops (input projection and the
// classifier GEMM) run on v_wmma_f32_16x16x32_f16 with f32 accumulation;
// fragment builds are branchless (selects, in-bounds loads for all lanes) so
// EXEC stays all-ones into every WMMA. The edge softmax/aggregation is an
// L2-resident (~35 MB << 192 MB) atomic scatter — bandwidth/atomic bound,
// not compute bound.

#define B_      64
#define NN_     1024
#define N_NODES (B_ * NN_)          // 65536
#define DIN_    10
#define DIM_    64
#define E_      20480
#define BE_     (B_ * E_)           // 1310720
#define TOTE_   (BE_ + N_NODES)     // 1376256 (edges + self loops)
#define EPS_    1e-5f

typedef __attribute__((ext_vector_type(16))) _Float16 v16h;
typedef __attribute__((ext_vector_type(8)))  float    v8f;

// ---------------------------------------------------------------- init
__global__ void tg_init(float* __restrict__ agg, float* __restrict__ denom,
                        float* __restrict__ m, float* __restrict__ stats) {
    size_t stride = (size_t)gridDim.x * blockDim.x;
    for (size_t i = (size_t)blockIdx.x * blockDim.x + threadIdx.x;
         i < (size_t)N_NODES * DIM_; i += stride) {
        agg[i] = 0.f;
        if (i < N_NODES) { denom[i] = 0.f; m[i] = -1.0e30f; }
        if (i < 256)     stats[i] = 0.f;
    }
}

// ------------------------------------------ x = data @ lin_W.T  (WMMA)
// One wave per 16-node tile. A: 16x32 f16 (K=10 zero-padded), B: 32x16 f16
// (B[k][n] = lin_W[n][k]), 4 N-tiles cover DIM=64.
// ISA 16-bit A layout: lanes 0-15 hold M=lane with halves e=0..7 -> K=e,
// e=8..15 -> K=16+(e-8); lanes 16-31 hold the same M with e=0..7 -> K=8+e.
// All slots with K >= 10 are zero. Branchless build: every lane loads its
// full 10-float row (always in-bounds) and selects the slot values.
__global__ void tg_lin_wmma(const float* __restrict__ data,
                            const float* __restrict__ linW,
                            const float* __restrict__ atti,
                            const float* __restrict__ attj,
                            float* __restrict__ x,
                            float* __restrict__ si,
                            float* __restrict__ sj) {
    __shared__ float tile[16][DIM_ + 1];   // +1 pad: avoid 16-way bank conflict
    const int  lane = threadIdx.x;         // 0..31
    const bool hi   = lane >= 16;
    const int  l16  = lane & 15;
    const int  row0 = blockIdx.x * 16;
    const int  row  = row0 + l16;

    // A fragment (branchless)
    v16h a;
    const float* dr = data + (size_t)row * DIN_;
#pragma unroll
    for (int e = 0; e < 8; ++e) {
        const float lo = dr[e];                            // K=e      (lanes 0-15)
        const float hv = (e < 2) ? dr[8 + e] : 0.f;        // K=8+e    (lanes 16-31)
        a[e]     = (_Float16)(hi ? hv : lo);
        a[8 + e] = (_Float16)0.f;                          // K >= 16 pad
    }

    v8f acc[4];
#pragma unroll
    for (int nb = 0; nb < 4; ++nb) {
        const float* wc = linW + (size_t)(nb * 16 + l16) * DIN_; // lin_W[n][*]
        v16h b;
#pragma unroll
        for (int e = 0; e < 8; ++e) {
            const float lo = wc[e];
            const float hv = (e < 2) ? wc[8 + e] : 0.f;
            b[e]     = (_Float16)(hi ? hv : lo);
            b[8 + e] = (_Float16)0.f;
        }
        v8f c = {};
        c = __builtin_amdgcn_wmma_f32_16x16x32_f16(false, a, false, b,
                                                   (short)0, c, false, false);
        acc[nb] = c;
    }

    // D layout: vgpr r holds D[M][n], M = r + (hi?8:0), n = nb*16 + l16
#pragma unroll
    for (int nb = 0; nb < 4; ++nb) {
#pragma unroll
        for (int r = 0; r < 8; ++r) {
            const int M = r + (hi ? 8 : 0);
            const int n = nb * 16 + l16;
            const float v = acc[nb][r];
            tile[M][n] = v;
            x[(size_t)(row0 + M) * DIM_ + n] = v;
        }
    }
    __syncthreads();

    // attention scores from the LDS tile (lanes 0..15, one node each)
    if (!hi) {
        float a_i = 0.f, a_j = 0.f;
#pragma unroll
        for (int ch = 0; ch < DIM_; ++ch) {
            const float v = tile[l16][ch];
            a_i += v * atti[ch];
            a_j += v * attj[ch];
        }
        si[row] = a_i;
        sj[row] = a_j;
    }
}

// ---------------------------------------------------------------- edges
__device__ __forceinline__ void edge_sd(const int* __restrict__ ei, int idx,
                                        int& s, int& d) {
    if (idx < BE_) {
        const int g = idx / E_;
        const int e = idx - g * E_;
        s = ei[e]      + g * NN_;
        d = ei[E_ + e] + g * NN_;
    } else {
        s = d = idx - BE_;
    }
}

__device__ __forceinline__ float leaky02(float v) { return v > 0.f ? v : 0.2f * v; }

__global__ void tg_edge_max(const int* __restrict__ ei,
                            const float* __restrict__ si,
                            const float* __restrict__ sj,
                            float* __restrict__ m) {
    const int idx = blockIdx.x * blockDim.x + threadIdx.x;
    if (idx >= TOTE_) return;
    int s, d; edge_sd(ei, idx, s, d);
    const float l = leaky02(si[d] + sj[s]);
    unsigned* um = (unsigned*)(m + d);
    unsigned old = *um;
    while (__uint_as_float(old) < l) {
        const unsigned prev = atomicCAS(um, old, __float_as_uint(l));
        if (prev == old) break;
        old = prev;
    }
}

__global__ void tg_edge_sum(const int* __restrict__ ei,
                            const float* __restrict__ si,
                            const float* __restrict__ sj,
                            const float* __restrict__ m,
                            float* __restrict__ denom) {
    const int idx = blockIdx.x * blockDim.x + threadIdx.x;
    if (idx >= TOTE_) return;
    int s, d; edge_sd(ei, idx, s, d);
    const float l = leaky02(si[d] + sj[s]);
    atomicAdd(denom + d, __expf(l - m[d]));
}

// 256 threads = 4 edges x 64 channels; alpha recomputed per thread (broadcast loads)
__global__ void tg_edge_agg(const int* __restrict__ ei,
                            const float* __restrict__ si,
                            const float* __restrict__ sj,
                            const float* __restrict__ m,
                            const float* __restrict__ denom,
                            const float* __restrict__ x,
                            float* __restrict__ agg) {
    const int eidx = blockIdx.x * 4 + (threadIdx.x >> 6);
    const int c    = threadIdx.x & 63;
    if (eidx >= TOTE_) return;
    int s, d; edge_sd(ei, eidx, s, d);
    const float l     = leaky02(si[d] + sj[s]);
    const float alpha = __expf(l - m[d]) / denom[d];
    atomicAdd(agg + (size_t)d * DIM_ + c, x[(size_t)s * DIM_ + c] * alpha);
}

// ----------------------------------------------------- BN stats / apply
// gnn_bias cancels exactly under BN1's mean subtraction, so raw moments of
// agg suffice (variance is shift-invariant).
__global__ void tg_stats1(const float* __restrict__ agg, float* __restrict__ stats) {
    const int c = threadIdx.x & 63;
    const int g = threadIdx.x >> 6;           // 0..3
    const int row0 = blockIdx.x * 1024;
    float s = 0.f, q = 0.f;
    for (int r = g; r < 1024; r += 4) {
        const float v = agg[(size_t)(row0 + r) * DIM_ + c];
        s += v; q += v * v;
    }
    atomicAdd(stats + c, s);
    atomicAdd(stats + 64 + c, q);
}

__global__ void tg_bn1(float* __restrict__ agg, float* __restrict__ stats,
                       const float* __restrict__ sc, const float* __restrict__ bi) {
    const float invN = 1.0f / (float)N_NODES;
    const int c = threadIdx.x & 63;
    const int g = threadIdx.x >> 6;
    const float mean  = stats[c] * invN;
    const float var   = stats[64 + c] * invN - mean * mean;
    const float rs    = rsqrtf(var + EPS_);
    const float scale = sc[c] * rs;
    const float shift = bi[c] - mean * scale;
    const int row0 = blockIdx.x * 1024;
    float s2 = 0.f, q2 = 0.f;
    for (int r = g; r < 1024; r += 4) {
        const size_t i = (size_t)(row0 + r) * DIM_ + c;
        float h = agg[i] * scale + shift;
        h = h > 0.f ? h : 0.f;                // ReLU
        agg[i] = h;
        s2 += h; q2 += h * h;
    }
    atomicAdd(stats + 128 + c, s2);
    atomicAdd(stats + 192 + c, q2);
}

// BN2 + ReLU + OutLayer dot -> y[node]
__global__ void tg_node_out(const float* __restrict__ agg,
                            const float* __restrict__ stats,
                            const float* __restrict__ sc,
                            const float* __restrict__ bi,
                            const float* __restrict__ outW,
                            const float* __restrict__ outb,
                            float* __restrict__ y) {
    const float invN = 1.0f / (float)N_NODES;
    const int node = blockIdx.x * blockDim.x + threadIdx.x;
    if (node >= N_NODES) return;
    float acc = 0.f;
#pragma unroll
    for (int c = 0; c < DIM_; ++c) {
        const float mu  = stats[128 + c] * invN;
        const float var = stats[192 + c] * invN - mu * mu;
        const float rs  = rsqrtf(var + EPS_);
        float h = (agg[(size_t)node * DIM_ + c] - mu) * rs * sc[c] + bi[c];
        h = h > 0.f ? h : 0.f;
        acc += h * outW[c];
    }
    y[node] = acc + outb[0];
}

// ------------------------------------------ classifier GEMM via WMMA
// out[b][k] = y[b,:] . cls_W[k,:] + cls_b[k];  [64,1024] x [1024,20].
// 8 waves: 4 M-tiles (batch rows) x 2 N-tiles (classes; cols 20..31 padded
// with zero via value selects on safely-addressed loads). K=1024 in 32 steps.
// Per-lane A halves at K-step kk: base = row*1024 + kk + (hi?8:0);
// e=0..7 -> base+e, e=8..15 -> base+16+(e-8) (per the 16-bit A/B VGPR layout).
__global__ void tg_cls_wmma(const float* __restrict__ y,
                            const float* __restrict__ clsW,
                            const float* __restrict__ clsb,
                            float* __restrict__ out) {
    const int  lane = threadIdx.x;            // 0..31
    const bool hi   = lane >= 16;
    const int  l16  = lane & 15;
    const int  m0   = (blockIdx.x >> 1) * 16;           // batch-row tile base
    const int  n    = (blockIdx.x & 1) * 16 + l16;      // class column
    const bool nv   = n < 20;

    const float* arow = y    + (size_t)(m0 + l16) * NN_ + (hi ? 8 : 0);
    const float* brow = clsW + (size_t)(nv ? n : 0) * NN_ + (hi ? 8 : 0);

    v8f c = {};
    for (int kk = 0; kk < NN_; kk += 32) {
        v16h a, b;
#pragma unroll
        for (int e = 0; e < 8; ++e) {
            a[e]     = (_Float16)arow[kk + e];
            a[8 + e] = (_Float16)arow[kk + 16 + e];
            const float b0 = brow[kk + e];          // always-safe address
            const float b1 = brow[kk + 16 + e];
            b[e]     = (_Float16)(nv ? b0 : 0.f);   // zero-pad cols 20..31
            b[8 + e] = (_Float16)(nv ? b1 : 0.f);
        }
        c = __builtin_amdgcn_wmma_f32_16x16x32_f16(false, a, false, b,
                                                   (short)0, c, false, false);
    }

    if (nv) {
        const float bias = clsb[n];
#pragma unroll
        for (int r = 0; r < 8; ++r) {
            const int M = m0 + r + (hi ? 8 : 0);    // batch index
            out[M * 20 + n] = c[r] + bias;          // [B,20] row-major
        }
    }
}

// ---------------------------------------------------------------- launch
extern "C" void kernel_launch(void* const* d_in, const int* in_sizes, int n_in,
                              void* d_out, int out_size, void* d_ws, size_t ws_size,
                              hipStream_t stream) {
    const float* data = (const float*)d_in[0];
    const int*   ei   = (const int*)  d_in[1];
    const float* linW = (const float*)d_in[2];
    const float* atti = (const float*)d_in[3];
    const float* attj = (const float*)d_in[4];
    // d_in[5] = gnn_bias: cancels under BN1 (mean subtraction) -> unused
    const float* bn1s = (const float*)d_in[6];
    const float* bn1b = (const float*)d_in[7];
    const float* bn2s = (const float*)d_in[8];
    const float* bn2b = (const float*)d_in[9];
    const float* outW = (const float*)d_in[10];
    const float* outb = (const float*)d_in[11];
    const float* clsW = (const float*)d_in[12];
    const float* clsb = (const float*)d_in[13];
    float* out = (float*)d_out;

    // workspace layout (floats): ~35 MB total, fits in L2 (192 MB)
    float* ws    = (float*)d_ws;
    float* x     = ws;                              // N*64
    float* si    = x  + (size_t)N_NODES * DIM_;     // N
    float* sj    = si + N_NODES;                    // N
    float* m     = sj + N_NODES;                    // N
    float* denom = m  + N_NODES;                    // N
    float* agg   = denom + N_NODES;                 // N*64
    float* stats = agg + (size_t)N_NODES * DIM_;    // 256 (sum1,sq1,sum2,sq2)
    float* y     = stats + 256;                     // N

    tg_init    <<<2048, 256, 0, stream>>>(agg, denom, m, stats);
    tg_lin_wmma<<<N_NODES / 16, 32, 0, stream>>>(data, linW, atti, attj, x, si, sj);
    tg_edge_max<<<(TOTE_ + 255) / 256, 256, 0, stream>>>(ei, si, sj, m);
    tg_edge_sum<<<(TOTE_ + 255) / 256, 256, 0, stream>>>(ei, si, sj, m, denom);
    tg_edge_agg<<<(TOTE_ + 3) / 4, 256, 0, stream>>>(ei, si, sj, m, denom, x, agg);
    tg_stats1  <<<N_NODES / 1024, 256, 0, stream>>>(agg, stats);
    tg_bn1     <<<N_NODES / 1024, 256, 0, stream>>>(agg, stats, bn1s, bn1b);
    tg_node_out<<<N_NODES / 256, 256, 0, stream>>>(agg, stats, bn2s, bn2b, outW, outb, y);
    tg_cls_wmma<<<8, 32, 0, stream>>>(y, clsW, clsb, out);
}
